// StructuralGraphTransformer_31568009625746
// MI455X (gfx1250) — matrix operand order repaired and checked by reading, not compiled, not run on hardware
//
#include <hip/hip_runtime.h>

#define BDIM 4
#define SDIM 2048
#define EDIM 1024
#define HDIM 16
#define DDIM 64
#define MTOT (BDIM * SDIM) /* 8192 */
#define LNEPS 1e-5f

typedef __attribute__((ext_vector_type(16))) __bf16 v16bf;
typedef __attribute__((ext_vector_type(8)))  __bf16 v8bf;
typedef __attribute__((ext_vector_type(8)))  float  v8f;
typedef __attribute__((ext_vector_type(4)))  unsigned int v4u;
typedef __attribute__((ext_vector_type(8)))  int v8i;
typedef __attribute__((ext_vector_type(4)))  int v4i;

#if __has_builtin(__builtin_amdgcn_tensor_load_to_lds) && __has_builtin(__builtin_amdgcn_s_wait_tensorcnt)
#define HAVE_TDM 1
#else
#define HAVE_TDM 0
#endif

union BF16x16 { v16bf v; v8bf h[2]; };

__device__ inline __bf16 f2bf(float f) {
  unsigned u = __builtin_bit_cast(unsigned, f);
  unsigned r = (u + 0x7FFFu + ((u >> 16) & 1u)) >> 16;  // round-to-nearest-even
  unsigned short s = (unsigned short)r;
  return __builtin_bit_cast(__bf16, s);
}

// 16-bit A-matrix 16x32 (MxK): lane&15 = row M; lanes 0-15 hold K {k0..k0+7, k0+16..k0+23},
// lanes 16-31 hold K {k0+8..k0+15, k0+24..k0+31}. Source row-major [M][K], leading dim ld.
__device__ inline v16bf load_fragA(const __bf16* base, int ld, int m0, int k0, int lane) {
  int row = m0 + (lane & 15);
  int kb  = k0 + ((lane >> 4) << 3);
  BF16x16 u;
  u.h[0] = *(const v8bf*)(base + (size_t)row * ld + kb);
  u.h[1] = *(const v8bf*)(base + (size_t)row * ld + kb + 16);
  return u.v;
}

// 16-bit B-matrix 32x16 (KxN): lane&15 = col N; lanes 0-15 hold K {k0..k0+7, k0+16..k0+23},
// lanes 16-31 hold K {k0+8..k0+15, k0+24..k0+31}. Source "column-major" [N][K], leading dim ld.
__device__ inline v16bf load_fragB(const __bf16* base, int ld, int n0, int k0, int lane) {
  int col = n0 + (lane & 15);
  int kb  = k0 + ((lane >> 4) << 3);
  BF16x16 u;
  u.h[0] = *(const v8bf*)(base + (size_t)col * ld + kb);
  u.h[1] = *(const v8bf*)(base + (size_t)col * ld + kb + 16);
  return u.v;
}

__device__ inline v8f wmma_bf16(v16bf a, v16bf b, v8f c) {
  return __builtin_amdgcn_wmma_f32_16x16x32_bf16(false, a, false, b, (short)0, c, false, false);
}

#if HAVE_TDM
// TDM: 2D tile (32 x 16 rows) of 8-byte elements from a [2048 x 2048] int64 tensor -> LDS.
// D# per CDNA5 ISA ch.8: group0 = count/lds_addr/global_addr/type, group1 = dims+strides.
__device__ inline void tdm_load_tile(const long long* tile_ptr, unsigned lds_off) {
  unsigned long long ga = (unsigned long long)(uintptr_t)tile_ptr;
  v4u g0;
  g0[0] = 1u;                                               // count=1, user mode
  g0[1] = lds_off;                                          // lds_addr (bytes)
  g0[2] = (unsigned)(ga & 0xFFFFFFFFu);                     // global_addr[31:0]
  g0[3] = (unsigned)((ga >> 32) & 0x01FFFFFFu) | (2u << 30); // addr[56:32] | type=2
  unsigned long long q0 = (3ull << 16)                      // data_size=3 (8B)
                        | ((unsigned long long)(SDIM & 0xFFFF) << 48); // tensor_dim0 lo16
  unsigned long long q1 = ((unsigned long long)(SDIM >> 16) & 0xFFFFull) // tensor_dim0 hi16
                        | ((unsigned long long)SDIM << 16)  // tensor_dim1 @bit80
                        | (32ull << 48);                    // tile_dim0=32 @bit112
  unsigned long long q2 = 16ull                             // tile_dim1=16 @bit128
                        | ((unsigned long long)SDIM << 32); // tensor_dim0_stride lo32 @bit160
  unsigned long long q3 = 0;                                // stride hi16=0, dim1_stride=0
  v8i g1;
  g1[0] = (int)(unsigned)(q0 & 0xFFFFFFFFu); g1[1] = (int)(unsigned)(q0 >> 32);
  g1[2] = (int)(unsigned)(q1 & 0xFFFFFFFFu); g1[3] = (int)(unsigned)(q1 >> 32);
  g1[4] = (int)(unsigned)(q2 & 0xFFFFFFFFu); g1[5] = (int)(unsigned)(q2 >> 32);
  g1[6] = (int)(unsigned)(q3 & 0xFFFFFFFFu); g1[7] = (int)(unsigned)(q3 >> 32);
  v4i gz = {0, 0, 0, 0};
#if defined(__clang_major__) && (__clang_major__ >= 23)
  v8i gz8 = {0, 0, 0, 0, 0, 0, 0, 0};
  __builtin_amdgcn_tensor_load_to_lds(g0, g1, gz, gz, gz8, 0);
#else
  __builtin_amdgcn_tensor_load_to_lds(g0, g1, gz, gz, 0);
#endif
}
#endif

// ---------------------------------------------------------------- converts
__global__ void k_cvt(const float* __restrict__ src, __bf16* __restrict__ dst, int n) {
  int i = blockIdx.x * blockDim.x + threadIdx.x;
  if (i < n) dst[i] = f2bf(src[i]);
}

// ---------------------------------------------------------------- QKV GEMM
// Ping-pong software pipeline: two fragment sets, k-loop unrolled x2, last step peeled
// -> no register rotation copies, no branches in steady state.
__global__ __launch_bounds__(256) void k_qkv(
    const __bf16* __restrict__ xb,
    const __bf16* __restrict__ wq, const __bf16* __restrict__ wk, const __bf16* __restrict__ wv,
    const float* __restrict__ bq, const float* __restrict__ bk, const float* __restrict__ bv,
    __bf16* __restrict__ Qb, __bf16* __restrict__ Kb, __bf16* __restrict__ Vt) {
  const int lane = threadIdx.x & 31;
  const int wave = threadIdx.x >> 5;
  const int m0  = blockIdx.x * 16;
  const int nv0 = (blockIdx.y * 8 + wave) * 64;   // 0..3071
  const int mat = nv0 >> 10;                      // 0=Q 1=K 2=V
  const int nc0 = nv0 & 1023;
  const __bf16* W    = (mat == 0) ? wq : (mat == 1) ? wk : wv;
  const float*  bias = (mat == 0) ? bq : (mat == 1) ? bk : bv;

  v8f acc[4];
  acc[0] = {}; acc[1] = {}; acc[2] = {}; acc[3] = {};

  v16bf a0 = load_fragA(xb, EDIM, m0, 0, lane);
  v16bf b0[4], b1[4];
#pragma unroll
  for (int t = 0; t < 4; ++t) b0[t] = load_fragB(W, EDIM, nc0 + t * 16, 0, lane);
  v16bf a1;

  for (int k0 = 0; k0 < EDIM - 64; k0 += 64) {
    a1 = load_fragA(xb, EDIM, m0, k0 + 32, lane);
#pragma unroll
    for (int t = 0; t < 4; ++t) b1[t] = load_fragB(W, EDIM, nc0 + t * 16, k0 + 32, lane);
#pragma unroll
    for (int t = 0; t < 4; ++t) acc[t] = wmma_bf16(a0, b0[t], acc[t]);
    a0 = load_fragA(xb, EDIM, m0, k0 + 64, lane);
#pragma unroll
    for (int t = 0; t < 4; ++t) b0[t] = load_fragB(W, EDIM, nc0 + t * 16, k0 + 64, lane);
#pragma unroll
    for (int t = 0; t < 4; ++t) acc[t] = wmma_bf16(a1, b1[t], acc[t]);
  }
  // epilogue: last two K-steps (EDIM-64, EDIM-32)
  a1 = load_fragA(xb, EDIM, m0, EDIM - 32, lane);
#pragma unroll
  for (int t = 0; t < 4; ++t) b1[t] = load_fragB(W, EDIM, nc0 + t * 16, EDIM - 32, lane);
#pragma unroll
  for (int t = 0; t < 4; ++t) acc[t] = wmma_bf16(a0, b0[t], acc[t]);
#pragma unroll
  for (int t = 0; t < 4; ++t) acc[t] = wmma_bf16(a1, b1[t], acc[t]);

  const int b  = m0 / SDIM;
  const int h  = nc0 >> 6;
  const int bh = b * HDIM + h;
#pragma unroll
  for (int t = 0; t < 4; ++t) {
    const int dcol = (nc0 & 63) + t * 16 + (lane & 15);
    const float bval = bias[nc0 + t * 16 + (lane & 15)];
#pragma unroll
    for (int r = 0; r < 8; ++r) {
      int row = m0 + r + ((lane >> 4) << 3);
      int s   = row - b * SDIM;
      __bf16 o = f2bf(acc[t][r] + bval);
      if (mat == 2) {
        Vt[((size_t)bh * DDIM + dcol) * SDIM + s] = o;
      } else {
        __bf16* dst = (mat == 0) ? Qb : Kb;
        dst[((size_t)bh * SDIM + s) * DDIM + dcol] = o;
      }
    }
  }
}

// ---------------------------------------------------------------- flash attention
// Block: 512 threads = 16 waves = 16 heads of one (b, 16-query-row) slice.
// Distances tiles double-buffered in LDS via TDM; K/V fragments hoisted so global
// loads overlap score WMMAs + softmax VALU work.
__global__ __launch_bounds__(512) void k_attn(
    const __bf16* __restrict__ Qb, const __bf16* __restrict__ Kb, const __bf16* __restrict__ Vt,
    const long long* __restrict__ dist, const float* __restrict__ dist_bias,
    __bf16* __restrict__ Ob) {
  __shared__ __align__(16) long long d_lds[2][16][32];     // 8 KB distance tiles
  __shared__ __align__(16) __bf16 p_lds[16][16][32];       // per-wave P transpose staging

  const int lane = threadIdx.x & 31;
  const int wave = threadIdx.x >> 5;          // head
  const int s0 = blockIdx.x * 16;
  const int b  = blockIdx.y;
  const int bh = b * HDIM + wave;

  const __bf16* Qp = Qb + (size_t)bh * SDIM * DDIM;
  const __bf16* Kp = Kb + (size_t)bh * SDIM * DDIM;
  const __bf16* Vp = Vt + (size_t)bh * DDIM * SDIM;
  const long long* drow = dist + ((size_t)b * SDIM + s0) * SDIM;   // tile row base

  const float db0 = dist_bias[0], db1 = dist_bias[1], db2 = dist_bias[2],
              db3 = dist_bias[3], db4 = dist_bias[4];
  const float BL2 = 1.4426950408889634f;      // log2(e)
  const float SCL = 0.125f * BL2;             // (1/sqrt(64)) * log2(e)

  const v16bf qa0 = load_fragA(Qp, DDIM, s0, 0,  lane);
  const v16bf qa1 = load_fragA(Qp, DDIM, s0, 32, lane);

  v8f acc[4];
  acc[0] = {}; acc[1] = {}; acc[2] = {}; acc[3] = {};
  float mrow[8], lrow[8];
#pragma unroll
  for (int r = 0; r < 8; ++r) { mrow[r] = -1e30f; lrow[r] = 0.0f; }

  // prologue: bring tile 0 into buffer 0
#if HAVE_TDM
  if (wave == 0) tdm_load_tile(drow, (unsigned)(uintptr_t)&d_lds[0][0][0]);
#else
  d_lds[0][threadIdx.x >> 5][threadIdx.x & 31] = drow[(size_t)(threadIdx.x >> 5) * SDIM + (threadIdx.x & 31)];
#endif

  for (int t0 = 0; t0 < SDIM; t0 += 32) {
    const int cur = (t0 >> 5) & 1;
#if HAVE_TDM
    if (wave == 0) __builtin_amdgcn_s_wait_tensorcnt(0);   // tile(t0) landed in LDS
#endif
    __syncthreads();   // publish tile(t0); all prior readers of the other buffer are done
#if HAVE_TDM
    if (wave == 0 && t0 + 32 < SDIM)
      tdm_load_tile(drow + t0 + 32, (unsigned)(uintptr_t)&d_lds[cur ^ 1][0][0]);
#else
    if (t0 + 32 < SDIM)
      d_lds[cur ^ 1][threadIdx.x >> 5][threadIdx.x & 31] =
          drow[(size_t)(threadIdx.x >> 5) * SDIM + (t0 + 32 + (threadIdx.x & 31))];
#endif

    // hoist all global fragments for this iteration: 16 b128 loads issued up front
    v16bf kb0 = load_fragB(Kp, DDIM, t0,      0,  lane);
    v16bf kb1 = load_fragB(Kp, DDIM, t0,      32, lane);
    v16bf kb2 = load_fragB(Kp, DDIM, t0 + 16, 0,  lane);
    v16bf kb3 = load_fragB(Kp, DDIM, t0 + 16, 32, lane);
    v16bf vb0 = load_fragB(Vp, SDIM, 0,  t0, lane);
    v16bf vb1 = load_fragB(Vp, SDIM, 16, t0, lane);
    v16bf vb2 = load_fragB(Vp, SDIM, 32, t0, lane);
    v16bf vb3 = load_fragB(Vp, SDIM, 48, t0, lane);

    // scores for 32 keys: two 16x16 tiles
    v8f c0 = {}, c1 = {};
    c0 = wmma_bf16(qa0, kb0, c0);
    c0 = wmma_bf16(qa1, kb1, c0);
    c1 = wmma_bf16(qa0, kb2, c1);
    c1 = wmma_bf16(qa1, kb3, c1);

    const int colL = lane & 15;
#pragma unroll
    for (int r = 0; r < 8; ++r) {
      int row = r + ((lane >> 4) << 3);
      int dL = (int)d_lds[cur][row][colL];
      int dH = (int)d_lds[cur][row][colL + 16];
      dL = dL < 0 ? 0 : (dL > 4 ? 4 : dL);
      dH = dH < 0 ? 0 : (dH > 4 ? 4 : dH);
      float bL = dL == 0 ? db0 : dL == 1 ? db1 : dL == 2 ? db2 : dL == 3 ? db3 : db4;
      float bH = dH == 0 ? db0 : dH == 1 ? db1 : dH == 2 ? db2 : dH == 3 ? db3 : db4;
      float sL = c0[r] * SCL + bL * BL2;
      float sH = c1[r] * SCL + bH * BL2;
      float mx = fmaxf(sL, sH);   // row-max: butterfly inside 16-lane halves (distinct rows)
      mx = fmaxf(mx, __shfl_xor(mx, 1, 32));
      mx = fmaxf(mx, __shfl_xor(mx, 2, 32));
      mx = fmaxf(mx, __shfl_xor(mx, 4, 32));
      mx = fmaxf(mx, __shfl_xor(mx, 8, 32));
      float mnew  = fmaxf(mrow[r], mx);
      float alpha = exp2f(mrow[r] - mnew);
      float pL = exp2f(sL - mnew);
      float pH = exp2f(sH - mnew);
      float ps = pL + pH;
      ps += __shfl_xor(ps, 1, 32);
      ps += __shfl_xor(ps, 2, 32);
      ps += __shfl_xor(ps, 4, 32);
      ps += __shfl_xor(ps, 8, 32);
      lrow[r] = lrow[r] * alpha + ps;
      mrow[r] = mnew;
#pragma unroll
      for (int t = 0; t < 4; ++t) acc[t][r] *= alpha;
      p_lds[wave][row][colL]      = f2bf(pL);   // stage P (C-layout -> A-layout via LDS)
      p_lds[wave][row][colL + 16] = f2bf(pH);
    }

    // P(16x32) @ V(32x64): per-wave LDS RAW kept in order by DScnt; V frags already resident
    v16bf pa = load_fragA(&p_lds[wave][0][0], 32, 0, 0, lane);
    acc[0] = wmma_bf16(pa, vb0, acc[0]);
    acc[1] = wmma_bf16(pa, vb1, acc[1]);
    acc[2] = wmma_bf16(pa, vb2, acc[2]);
    acc[3] = wmma_bf16(pa, vb3, acc[3]);
  }

  // normalize and store context at [b,s][h*64+d]
#pragma unroll
  for (int t = 0; t < 4; ++t) {
#pragma unroll
    for (int r = 0; r < 8; ++r) {
      int row = r + ((lane >> 4) << 3);
      float o = acc[t][r] / lrow[r];
      int s   = s0 + row;
      int col = wave * DDIM + t * 16 + (lane & 15);
      Ob[((size_t)(b * SDIM + s)) * EDIM + col] = f2bf(o);
    }
  }
}

// ---------------------------------------------------------------- output projection (ping-pong)
__global__ __launch_bounds__(256) void k_oproj(
    const __bf16* __restrict__ Ob, const __bf16* __restrict__ wo,
    const float* __restrict__ bo, float* __restrict__ Yf) {
  const int lane = threadIdx.x & 31;
  const int wave = threadIdx.x >> 5;
  const int m0 = blockIdx.x * 16;
  const int n0 = (blockIdx.y * 8 + wave) * 64;

  v8f acc[4];
  acc[0] = {}; acc[1] = {}; acc[2] = {}; acc[3] = {};

  v16bf a0 = load_fragA(Ob, EDIM, m0, 0, lane);
  v16bf b0[4], b1[4];
#pragma unroll
  for (int t = 0; t < 4; ++t) b0[t] = load_fragB(wo, EDIM, n0 + t * 16, 0, lane);
  v16bf a1;

  for (int k0 = 0; k0 < EDIM - 64; k0 += 64) {
    a1 = load_fragA(Ob, EDIM, m0, k0 + 32, lane);
#pragma unroll
    for (int t = 0; t < 4; ++t) b1[t] = load_fragB(wo, EDIM, n0 + t * 16, k0 + 32, lane);
#pragma unroll
    for (int t = 0; t < 4; ++t) acc[t] = wmma_bf16(a0, b0[t], acc[t]);
    a0 = load_fragA(Ob, EDIM, m0, k0 + 64, lane);
#pragma unroll
    for (int t = 0; t < 4; ++t) b0[t] = load_fragB(wo, EDIM, n0 + t * 16, k0 + 64, lane);
#pragma unroll
    for (int t = 0; t < 4; ++t) acc[t] = wmma_bf16(a1, b1[t], acc[t]);
  }
  a1 = load_fragA(Ob, EDIM, m0, EDIM - 32, lane);
#pragma unroll
  for (int t = 0; t < 4; ++t) b1[t] = load_fragB(wo, EDIM, n0 + t * 16, EDIM - 32, lane);
#pragma unroll
  for (int t = 0; t < 4; ++t) acc[t] = wmma_bf16(a0, b0[t], acc[t]);
#pragma unroll
  for (int t = 0; t < 4; ++t) acc[t] = wmma_bf16(a1, b1[t], acc[t]);

#pragma unroll
  for (int t = 0; t < 4; ++t) {
    int col = n0 + t * 16 + (lane & 15);
    float bval = bo[col];
#pragma unroll
    for (int r = 0; r < 8; ++r) {
      int row = m0 + r + ((lane >> 4) << 3);
      Yf[(size_t)row * EDIM + col] = acc[t][r] + bval;
    }
  }
}

// ---------------------------------------------------------------- residual + layernorm
__global__ __launch_bounds__(256) void k_ln(
    const float* __restrict__ x, const float* __restrict__ Yf,
    const float* __restrict__ g, const float* __restrict__ bb,
    float* __restrict__ out) {
  const int row  = blockIdx.x;
  const int lane = threadIdx.x & 31;
  const int wave = threadIdx.x >> 5;
  const float* xr = x  + (size_t)row * EDIM;
  const float* yr = Yf + (size_t)row * EDIM;
  __shared__ float red[8];

  float v[4];
  float sum = 0.0f;
#pragma unroll
  for (int i = 0; i < 4; ++i) {
    int e = threadIdx.x + i * 256;
    v[i] = xr[e] + yr[e];
    sum += v[i];
  }
  for (int m = 1; m < 32; m <<= 1) sum += __shfl_xor(sum, m, 32);
  if (lane == 0) red[wave] = sum;
  __syncthreads();
  float tot = 0.0f;
#pragma unroll
  for (int w = 0; w < 8; ++w) tot += red[w];
  float mu = tot * (1.0f / EDIM);

  float vs = 0.0f;
#pragma unroll
  for (int i = 0; i < 4; ++i) { float dlt = v[i] - mu; vs += dlt * dlt; }
  for (int m = 1; m < 32; m <<= 1) vs += __shfl_xor(vs, m, 32);
  __syncthreads();
  if (lane == 0) red[wave] = vs;
  __syncthreads();
  float tv = 0.0f;
#pragma unroll
  for (int w = 0; w < 8; ++w) tv += red[w];
  float rstd = rsqrtf(tv * (1.0f / EDIM) + LNEPS);

#pragma unroll
  for (int i = 0; i < 4; ++i) {
    int e = threadIdx.x + i * 256;
    out[(size_t)row * EDIM + e] = (v[i] - mu) * rstd * g[e] + bb[e];
  }
}

// ---------------------------------------------------------------- launcher
extern "C" void kernel_launch(void* const* d_in, const int* in_sizes, int n_in,
                              void* d_out, int out_size, void* d_ws, size_t ws_size,
                              hipStream_t stream) {
  const float*     x    = (const float*)d_in[0];
  const long long* dist = (const long long*)d_in[1];   // jnp.int64
  const float* Wq = (const float*)d_in[2];
  const float* bq = (const float*)d_in[3];
  const float* Wk = (const float*)d_in[4];
  const float* bk = (const float*)d_in[5];
  const float* Wv = (const float*)d_in[6];
  const float* bv = (const float*)d_in[7];
  const float* Wo = (const float*)d_in[8];
  const float* bo = (const float*)d_in[9];
  const float* dist_bias = (const float*)d_in[10];
  const float* ln_g = (const float*)d_in[11];
  const float* ln_b = (const float*)d_in[12];
  float* out = (float*)d_out;

  char* ws = (char*)d_ws;
  size_t off = 0;
  const size_t NXE = (size_t)MTOT * EDIM;
  const size_t NWW = (size_t)EDIM * EDIM;
  __bf16* xb  = (__bf16*)(ws + off); off += NXE * 2;
  __bf16* wqb = (__bf16*)(ws + off); off += NWW * 2;
  __bf16* wkb = (__bf16*)(ws + off); off += NWW * 2;
  __bf16* wvb = (__bf16*)(ws + off); off += NWW * 2;
  __bf16* wob = (__bf16*)(ws + off); off += NWW * 2;
  __bf16* Qb  = (__bf16*)(ws + off); off += NXE * 2;   // [bh][s][d]
  __bf16* Kb  = (__bf16*)(ws + off); off += NXE * 2;   // [bh][s][d]
  __bf16* Vt  = (__bf16*)(ws + off); off += NXE * 2;   // [bh][d][s]
  __bf16* Ob  = (__bf16*)(ws + off); off += NXE * 2;   // [b*s][h*d]
  float*  Yf  = (float*)(ws + off);  off += NXE * 4;   // proj out, pre-residual
  (void)ws_size; (void)in_sizes; (void)n_in; (void)out_size;

  k_cvt<<<(int)((NXE + 255) / 256), 256, 0, stream>>>(x,  xb,  (int)NXE);
  k_cvt<<<(int)((NWW + 255) / 256), 256, 0, stream>>>(Wq, wqb, (int)NWW);
  k_cvt<<<(int)((NWW + 255) / 256), 256, 0, stream>>>(Wk, wkb, (int)NWW);
  k_cvt<<<(int)((NWW + 255) / 256), 256, 0, stream>>>(Wv, wvb, (int)NWW);
  k_cvt<<<(int)((NWW + 255) / 256), 256, 0, stream>>>(Wo, wob, (int)NWW);

  dim3 gqkv(MTOT / 16, 6);
  k_qkv<<<gqkv, 256, 0, stream>>>(xb, wqb, wkb, wvb, bq, bk, bv, Qb, Kb, Vt);

  dim3 gattn(SDIM / 16, BDIM);
  k_attn<<<gattn, 512, 0, stream>>>(Qb, Kb, Vt, dist, dist_bias, Ob);

  dim3 gop(MTOT / 16, 2);
  k_oproj<<<gop, 256, 0, stream>>>(Ob, wob, bo, Yf);

  k_ln<<<MTOT, 256, 0, stream>>>(x, Yf, ln_g, ln_b, out);
}
